// SocialLSTM_33638183862687
// MI455X (gfx1250) — compile-verified, matrix operand order checked by reading, not run on hardware
//
#include <hip/hip_runtime.h>
#include <hip/hip_bf16.h>

// ---------------------------------------------------------------------------
// Social LSTM for MI455X (gfx1250, wave32, WMMA 16x16x32 bf16)
//
// Pipeline:
//   K1 prep_kernel      : repack weights to bf16 (Wcat=[W_ih|W_hh], WsT, WhoT), bsum
//   K2 ds_embed_kernel  : downsample (mean/max/min per 16x16 cell) + embed -> bf16
//   K3 lstm_kernel      : persistent; 16 WGs x 16 batches; full T=48 recurrence
//                         in LDS; all matmuls via v_wmma_f32_16x16x32_bf16
// ---------------------------------------------------------------------------

typedef __attribute__((ext_vector_type(16))) __bf16   v16bf;
typedef __attribute__((ext_vector_type(8)))  float    v8f;
typedef __attribute__((ext_vector_type(4)))  unsigned int u32x4;

union TileU { u32x4 u[2]; v16bf v; };

__device__ __forceinline__ unsigned short f2bf(float f) {
    unsigned int u = __float_as_uint(f);
    u += 0x7fffu + ((u >> 16) & 1u);        // round-to-nearest-even
    return (unsigned short)(u >> 16);
}

__device__ __forceinline__ float sigmoidf(float x) {
    return 1.0f / (1.0f + __expf(-x));
}

// Load a 16x32 bf16 A/B tile in WMMA register layout from a row-major matrix.
// lane 0-15 : row=lane,    K-chunks {0..7, 16..23}
// lane 16-31: row=lane-16, K-chunks {8..15,24..31}
__device__ __forceinline__ v16bf load_tile(const unsigned short* base, int ld) {
    const int lane = threadIdx.x & 31;
    const int r    = lane & 15;
    const int hi   = lane >> 4;
    const unsigned short* p = base + r * ld + hi * 8;
    TileU tu;
    tu.u[0] = *reinterpret_cast<const u32x4*>(p);
    tu.u[1] = *reinterpret_cast<const u32x4*>(p + 16);
    return tu.v;
}

__device__ __forceinline__ v8f wmma_bf16(v16bf a, v16bf b, v8f c) {
#if defined(__HIP_DEVICE_COMPILE__)
    return __builtin_amdgcn_wmma_f32_16x16x32_bf16(false, a, false, b,
                                                   (short)0, c, false, false);
#else
    return c;
#endif
}

// ---------------------------------------------------------------------------
// Problem constants
#define TT   48
#define BB   256
#define N2   16
#define EE   64
#define HH   128
#define GG   512            // 4*H gates
#define KK   256            // [frame(64) | social(64) | h(128)]
#define BT   16             // batch tile per workgroup

// Workspace layout (bytes)
#define WS_EMB   0u                                   // bf16 [T][16][B][64]  = 25165824 B
#define WS_WCAT  25165824u                            // bf16 [16][512][256]  =  4194304 B
#define WS_WST   29360128u                            // bf16 [64][2048]      =   262144 B
#define WS_WHOT  29622272u                            // bf16 [16][128][128]  =   524288 B
#define WS_BSUM  30146560u                            // f32  [16][512]       =    32768 B

// ---------------------------------------------------------------------------
__global__ void __launch_bounds__(256) prep_kernel(
    const float* __restrict__ W_ih, const float* __restrict__ b_ih,
    const float* __restrict__ W_hh, const float* __restrict__ b_hh,
    const float* __restrict__ Ws,   const float* __restrict__ W_ho,
    unsigned short* __restrict__ Wcat, unsigned short* __restrict__ WsT,
    unsigned short* __restrict__ WhoT, float* __restrict__ bsum)
{
    const int stride = gridDim.x * blockDim.x;
    const int idx0   = blockIdx.x * blockDim.x + threadIdx.x;

    // Wcat[n][g][k] = k<128 ? W_ih[n][g][k] : W_hh[n][g][k-128]
    for (int i = idx0; i < N2 * GG * KK; i += stride) {
        int k = i & 255, g = (i >> 8) & 511, nn = i >> 17;
        float v = (k < 128) ? W_ih[((size_t)nn * GG + g) * 128 + k]
                            : W_hh[((size_t)nn * GG + g) * 128 + (k - 128)];
        Wcat[i] = f2bf(v);
    }
    // WsT[e][j] = Ws[j][e]
    for (int i = idx0; i < EE * 2048; i += stride) {
        int j = i & 2047, e = i >> 11;
        WsT[i] = f2bf(Ws[(size_t)j * EE + e]);
    }
    // WhoT[n][o][h] = W_ho[n][h][o]
    for (int i = idx0; i < N2 * HH * HH; i += stride) {
        int h = i & 127, o = (i >> 7) & 127, nn = i >> 14;
        WhoT[i] = f2bf(W_ho[((size_t)nn * HH + h) * HH + o]);
    }
    // bsum = b_ih + b_hh
    for (int i = idx0; i < N2 * GG; i += stride) bsum[i] = b_ih[i] + b_hh[i];
}

// ---------------------------------------------------------------------------
// One block per (b,t) frame: 16 cells of 16x16 -> (mean,max,min) -> relu(. @ Wd + bd)
__global__ void __launch_bounds__(256) ds_embed_kernel(
    const float* __restrict__ x, const float* __restrict__ Wd,
    const float* __restrict__ bd, unsigned short* __restrict__ emb)
{
    __shared__ float ssum[256], smax[256], smin[256];
    __shared__ float cmean[16], cmx[16], cmn[16];

    const int fb = blockIdx.x;            // fb = b*48 + t (memory-linear frames)
    const int b  = fb / TT, t = fb % TT;
    const float* frame = x + (size_t)fb * 4096;

    const int tid  = threadIdx.x;
    const int cell = tid >> 4, r = tid & 15;
    const int ci = cell >> 2, cj = cell & 3;

    const float* rowp = frame + (ci * 16 + r) * 64 + cj * 16;
    float s = 0.f, mx = -3.4e38f, mn = 3.4e38f;
    #pragma unroll
    for (int k = 0; k < 16; ++k) {
        float v = rowp[k];
        s += v; mx = fmaxf(mx, v); mn = fminf(mn, v);
    }
    ssum[tid] = s; smax[tid] = mx; smin[tid] = mn;
    __syncthreads();
    for (int off = 8; off > 0; off >>= 1) {
        if (r < off) {
            ssum[tid] += ssum[tid + off];
            smax[tid]  = fmaxf(smax[tid], smax[tid + off]);
            smin[tid]  = fminf(smin[tid], smin[tid + off]);
        }
        __syncthreads();
    }
    if (r == 0) {
        cmean[cell] = ssum[tid] * (1.0f / 256.0f);
        cmx[cell] = smax[tid]; cmn[cell] = smin[tid];
    }
    __syncthreads();

    const float fm = cmean[cell], fx = cmx[cell], fn = cmn[cell];
    unsigned short* dst = emb + (((size_t)t * N2 + cell) * BB + b) * EE;
    #pragma unroll
    for (int q = 0; q < 4; ++q) {
        int e = (tid & 15) * 4 + q;
        float v = fm * Wd[e] + fx * Wd[64 + e] + fn * Wd[128 + e] + bd[e];
        v = v > 0.f ? v : 0.f;
        dst[e] = f2bf(v);
    }
}

// ---------------------------------------------------------------------------
// Persistent recurrent kernel. LDS (dynamic):
//   xt2  bf16 [16 cells][16 b][256 k]  (k: 0..63 frame | 64..127 social | 128..255 h)
//   cbuf f32  [16 cells][16 b][128]
//   red  f32  [16]
__global__ void __launch_bounds__(256, 1) lstm_kernel(
    const unsigned short* __restrict__ emb,
    const unsigned short* __restrict__ Wcat,
    const unsigned short* __restrict__ WsT,
    const unsigned short* __restrict__ WhoT,
    const float* __restrict__ bsum, const float* __restrict__ bs,
    const float* __restrict__ b_ho, const float* __restrict__ W_out,
    const float* __restrict__ b_out, float* __restrict__ y)
{
    extern __shared__ char smem[];
    unsigned short* xt2 = reinterpret_cast<unsigned short*>(smem);            // 131072 B
    float* cbuf         = reinterpret_cast<float*>(smem + 131072);            // 131072 B
    float* red          = reinterpret_cast<float*>(smem + 262144);            //     64 B

    const int B0   = blockIdx.x * BT;
    const int tid  = threadIdx.x;
    const int wave = tid >> 5;
    const int lane = tid & 31;
    const int colL = lane & 15;
    const int hi   = lane >> 4;

    // h = 0, c = 0
    for (int i = tid; i < N2 * BT * KK; i += 256) xt2[i] = 0;
    for (int i = tid; i < N2 * BT * HH; i += 256) cbuf[i] = 0.f;
    __syncthreads();

    for (int t = 0; t < TT; ++t) {
        // ---- stage frame embedding tile into xt2[:, :, 0:64] (bf16, 16B chunks)
        for (int idx = tid; idx < N2 * BT * EE / 8; idx += 256) {
            int n = idx >> 7, rem = idx & 127;
            int row = rem >> 3, c8 = (rem & 7) * 8;
            *reinterpret_cast<u32x4*>(xt2 + (n * BT + row) * KK + c8) =
                *reinterpret_cast<const u32x4*>(
                    emb + (((size_t)t * N2 + n) * BB + B0 + row) * EE + c8);
        }
        __syncthreads();

        // ---- social = relu(h_flat @ Ws + bs): M=16, K=2048, N=64 (waves 0..3)
        if (wave < 4) {
            v8f acc = {};
            for (int ks = 0; ks < 64; ++ks) {
                int n = ks >> 2, hoff = (ks & 3) * 32;
                v16bf a = load_tile(xt2 + n * BT * KK + 128 + hoff, KK);
                v16bf b = load_tile(WsT + (size_t)(wave * 16) * 2048 + ks * 32, 2048);
                acc = wmma_bf16(a, b, acc);
            }
            const int e   = wave * 16 + colL;
            const float bsv = bs[e];
            #pragma unroll
            for (int r = 0; r < 8; ++r) {
                int m = r + hi * 8;
                float v = acc[r] + bsv;
                v = v > 0.f ? v : 0.f;
                unsigned short hv = f2bf(v);
                #pragma unroll
                for (int n = 0; n < N2; ++n)
                    xt2[(n * BT + m) * KK + 64 + e] = hv;   // broadcast to all cells
            }
        }
        __syncthreads();

        // ---- per-cell: gates GEMM + LSTM pointwise + output head
        for (int n = 0; n < N2; ++n) {
            // gates: (16 x 256) @ (256 x 512); wave owns N-tiles {w, w+8, w+16, w+24}
            v8f acc_i = {}, acc_f = {}, acc_g = {}, acc_o = {};
            const unsigned short* wbase = Wcat + (size_t)n * GG * KK;
            for (int ks = 0; ks < 8; ++ks) {
                v16bf a  = load_tile(xt2 + n * BT * KK + ks * 32, KK);
                v16bf b0 = load_tile(wbase + (size_t)((wave +  0) * 16) * KK + ks * 32, KK);
                v16bf b1 = load_tile(wbase + (size_t)((wave +  8) * 16) * KK + ks * 32, KK);
                v16bf b2 = load_tile(wbase + (size_t)((wave + 16) * 16) * KK + ks * 32, KK);
                v16bf b3 = load_tile(wbase + (size_t)((wave + 24) * 16) * KK + ks * 32, KK);
                acc_i = wmma_bf16(a, b0, acc_i);
                acc_f = wmma_bf16(a, b1, acc_f);
                acc_g = wmma_bf16(a, b2, acc_g);
                acc_o = wmma_bf16(a, b3, acc_o);
            }
            __syncthreads();   // all gate-GEMM reads of xt2[n] h-slot complete

            const int gcol = wave * 16 + colL;            // column within each 128-chunk
            const float bi = bsum[n * GG + gcol];
            const float bf_ = bsum[n * GG + 128 + gcol];
            const float bg = bsum[n * GG + 256 + gcol];
            const float bo = bsum[n * GG + 384 + gcol];
            #pragma unroll
            for (int r = 0; r < 8; ++r) {
                int m = r + hi * 8;
                float iv = sigmoidf(acc_i[r] + bi);
                float fv = sigmoidf(acc_f[r] + bf_);
                float gv = tanhf(acc_g[r] + bg);
                float ov = sigmoidf(acc_o[r] + bo);
                float c_new = fv * cbuf[(n * BT + m) * HH + gcol] + iv * gv;
                cbuf[(n * BT + m) * HH + gcol] = c_new;
                float hval = ov * tanhf(c_new);
                xt2[(n * BT + m) * KK + 128 + gcol] = f2bf(hval);
            }
            __syncthreads();   // h_new visible for output head

            // output head: (16 x 128) @ (128 x 128); wave owns o-tile = wave
            v8f acco = {};
            for (int ks = 0; ks < 4; ++ks) {
                v16bf a = load_tile(xt2 + n * BT * KK + 128 + ks * 32, KK);
                v16bf b = load_tile(WhoT + ((size_t)n * HH + wave * 16) * HH + ks * 32, HH);
                acco = wmma_bf16(a, b, acco);
            }
            if (tid < 16) red[tid] = 0.f;
            __syncthreads();

            const int   ocol = wave * 16 + colL;
            const float bho  = b_ho[n * HH + ocol];
            const float wout = W_out[ocol];
            #pragma unroll
            for (int r = 0; r < 8; ++r) {
                int m = r + hi * 8;
                atomicAdd(&red[m], sigmoidf(acco[r] + bho) * wout);
            }
            __syncthreads();

            if (tid < 16)
                y[((size_t)t * BB + B0 + tid) * N2 + n] = red[tid] + b_out[0];
            __syncthreads();
        }
        __syncthreads();
    }
}

// ---------------------------------------------------------------------------
extern "C" void kernel_launch(void* const* d_in, const int* in_sizes, int n_in,
                              void* d_out, int out_size, void* d_ws, size_t ws_size,
                              hipStream_t stream) {
    const float* x     = (const float*)d_in[0];
    const float* Wd    = (const float*)d_in[1];
    const float* bd    = (const float*)d_in[2];
    const float* Ws    = (const float*)d_in[3];
    const float* bs    = (const float*)d_in[4];
    const float* W_ih  = (const float*)d_in[5];
    const float* b_ih  = (const float*)d_in[6];
    const float* W_hh  = (const float*)d_in[7];
    const float* b_hh  = (const float*)d_in[8];
    const float* W_ho  = (const float*)d_in[9];
    const float* b_ho  = (const float*)d_in[10];
    const float* W_out = (const float*)d_in[11];
    const float* b_out = (const float*)d_in[12];
    float* y = (float*)d_out;

    char* ws = (char*)d_ws;
    unsigned short* emb  = (unsigned short*)(ws + WS_EMB);
    unsigned short* Wcat = (unsigned short*)(ws + WS_WCAT);
    unsigned short* WsT  = (unsigned short*)(ws + WS_WST);
    unsigned short* WhoT = (unsigned short*)(ws + WS_WHOT);
    float*          bsum = (float*)(ws + WS_BSUM);

    prep_kernel<<<dim3(256), dim3(256), 0, stream>>>(
        W_ih, b_ih, W_hh, b_hh, Ws, W_ho, Wcat, WsT, WhoT, bsum);

    ds_embed_kernel<<<dim3(BB * TT), dim3(256), 0, stream>>>(x, Wd, bd, emb);

    const size_t smem_bytes = 131072 + 131072 + 64;   // xt2 + cbuf + red (262208 B)
    lstm_kernel<<<dim3(BB / BT), dim3(256), smem_bytes, stream>>>(
        emb, Wcat, WsT, WhoT, bsum, bs, b_ho, W_out, b_out, y);
}
